// MDECLayer_43611097924105
// MI455X (gfx1250) — matrix operand order repaired and checked by reading, not compiled, not run on hardware
//
#include <hip/hip_runtime.h>

typedef __attribute__((ext_vector_type(16))) __bf16 v16bf;
typedef __attribute__((ext_vector_type(8)))  float  v8f;

#define B_TOTAL 16384
#define DDIM 256
#define KC 32
#define WAVES_PER_WG 4
#define NTHREADS (WAVES_PER_WG * 32)                 // 128 (wave32)
#define ROWS_PER_WAVE 32                             // two 16-row A tiles
#define ROWS_PER_WG (ROWS_PER_WAVE * WAVES_PER_WG)   // 128
#define NTILES (KC * 16)                             // 512 (k, d-tile) tiles

#if __has_builtin(__builtin_amdgcn_global_load_async_to_lds_b128)
#define ASYNC_STAGE 1
// Builtin parameter type per hipcc diagnostic: int __vector_size__(16) *
typedef int v4i __attribute__((vector_size(16)));
typedef __attribute__((address_space(1))) v4i* gv4p;   // global src
typedef __attribute__((address_space(3))) v4i* lv4p;   // LDS dst
#endif

__device__ __forceinline__ void async_drain() {
#if defined(ASYNC_STAGE)
#if __has_builtin(__builtin_amdgcn_s_wait_asynccnt)
    __builtin_amdgcn_s_wait_asynccnt(0);
#else
    asm volatile("s_wait_asynccnt 0x0" ::: "memory");
#endif
#endif
}

// ---------------------------------------------------------------------------
// Prep 1: ST[k][d][e] = S[k][e][d], split into bf16 hi + bf16 lo residual.
// Tile (k,dt) is then a contiguous 8KB block -> linear staging copies.
// ---------------------------------------------------------------------------
__global__ void prep_split_transpose(const float* __restrict__ S,
                                     __bf16* __restrict__ STh,
                                     __bf16* __restrict__ STl) {
    size_t idx = (size_t)blockIdx.x * 256 + threadIdx.x;   // over [k][d][e]
    int e = idx & 255;
    size_t t = idx >> 8;
    int d = (int)(t & 255);
    int k = (int)(t >> 8);
    float v = S[((size_t)k * 256 + e) * 256 + d];
    __bf16 h = (__bf16)v;
    STh[idx] = h;
    STl[idx] = (__bf16)(v - (float)h);
}

// ---------------------------------------------------------------------------
// Prep 2: negu[k][d] = -((S_k + S_k^T) mu_k)[d],  c[k] = mu^T S mu
// ---------------------------------------------------------------------------
__global__ void prep_linear(const float* __restrict__ mu,
                            const float* __restrict__ S,
                            float* __restrict__ negu,
                            float* __restrict__ cvec) {
    int k = blockIdx.x;
    int d = threadIdx.x;                 // 256 threads
    __shared__ float smu[256];
    __shared__ float red[256];
    smu[d] = mu[k * 256 + d];
    __syncthreads();
    const float* Sk = S + (size_t)k * 65536;
    float rowdot = 0.f, coldot = 0.f;
    for (int e = 0; e < 256; ++e) {
        rowdot += Sk[d * 256 + e] * smu[e];
        coldot += Sk[e * 256 + d] * smu[e];
    }
    negu[k * 256 + d] = -(rowdot + coldot);
    red[d] = smu[d] * rowdot;
    __syncthreads();
    for (int s = 128; s > 0; s >>= 1) {
        if (d < s) red[d] += red[d + s];
        __syncthreads();
    }
    if (d == 0) cvec[k] = red[0];
}

// ---------------------------------------------------------------------------
// Main: per wave two resident 16-row A tiles (bf16 hi/lo). Double-buffered
// S^T slab in LDS staged via async global->LDS copies (ASYNCcnt); 4
// ds_load_b128 feed 6 WMMAs. Accumulator pre-loaded with -u_k (folds linear
// term in fp32), fused row-dot, wave32 shuffle reduce, cluster-normalize,
// emit q.
// ---------------------------------------------------------------------------
__global__ __launch_bounds__(NTHREADS)
void mdec_main(const float* __restrict__ X,
               const __bf16* __restrict__ STh,
               const __bf16* __restrict__ STl,
               const float* __restrict__ negu,
               const float* __restrict__ cvec,
               float* __restrict__ q) {
    __shared__ __bf16 sbh[2][16 * 256];              // 2 x 8 KB (hi)
    __shared__ __bf16 sbl[2][16 * 256];              // 2 x 8 KB (lo)
    __shared__ float  ts[ROWS_PER_WG][KC];           // 16 KB

    const int tid  = threadIdx.x;
    const int lane = tid & 31;
    const int wave = tid >> 5;
    const int lm   = lane & 15;          // A: row M / B,C,D: column N
    const int hi   = lane >> 4;          // lane half -> K sub-range
    const int rb0  = blockIdx.x * ROWS_PER_WG + wave * ROWS_PER_WAVE;
    const int rb1  = rb0 + 16;

    // ---- Resident A operands: bf16 hi/lo split of 32 rows (two 16-row tiles)
    // A 16x32 layout: lane&15 = M; lanes<16: K {0..7,16..23}, lanes>=16: +8.
    v16bf Ah0[8], Al0[8], Ah1[8], Al1[8];
    {
        const float* xr0 = X + (size_t)(rb0 + lm) * DDIM;
        const float* xr1 = X + (size_t)(rb1 + lm) * DDIM;
        #pragma unroll
        for (int es = 0; es < 8; ++es) {
            const int e0 = es * 32 + hi * 8;
            #pragma unroll
            for (int j = 0; j < 8; ++j) {
                float f0 = xr0[e0 + j],      f1 = xr0[e0 + 16 + j];
                float g0 = xr1[e0 + j],      g1 = xr1[e0 + 16 + j];
                __bf16 fh0 = (__bf16)f0, fh1 = (__bf16)f1;
                __bf16 gh0 = (__bf16)g0, gh1 = (__bf16)g1;
                Ah0[es][j] = fh0;  Ah0[es][8 + j] = fh1;
                Al0[es][j] = (__bf16)(f0 - (float)fh0);
                Al0[es][8 + j] = (__bf16)(f1 - (float)fh1);
                Ah1[es][j] = gh0;  Ah1[es][8 + j] = gh1;
                Al1[es][j] = (__bf16)(g0 - (float)gh0);
                Al1[es][8 + j] = (__bf16)(g1 - (float)gh1);
            }
        }
    }

#if defined(ASYNC_STAGE)
    // ---- staging: direct global -> LDS async copies, no VGPR round-trip
    auto stage = [&](int t, int b) {
        gv4p gh = (gv4p)(STh + ((size_t)t << 12));   // 4096 bf16 = 8 KB tile
        gv4p gl = (gv4p)(STl + ((size_t)t << 12));
        lv4p lh = (lv4p)&sbh[b][0];
        lv4p ll = (lv4p)&sbl[b][0];
        #pragma unroll
        for (int i = 0; i < 4; ++i) {
            const int idx = tid + i * NTHREADS;      // 512 b128 per array
            __builtin_amdgcn_global_load_async_to_lds_b128(gh + idx, lh + idx, 0, 0);
            __builtin_amdgcn_global_load_async_to_lds_b128(gl + idx, ll + idx, 0, 0);
        }
    };
    stage(0, 0);
    async_drain();
    __syncthreads();
#else
    // ---- staging fallback: global -> VGPR temps -> ds_store
    uint4 th[4], tl[4];
    auto stage_load = [&](int t) {
        const uint4* gh = (const uint4*)STh + ((size_t)t << 9);
        const uint4* gl = (const uint4*)STl + ((size_t)t << 9);
        #pragma unroll
        for (int i = 0; i < 4; ++i) {
            th[i] = gh[tid + i * NTHREADS];
            tl[i] = gl[tid + i * NTHREADS];
        }
    };
    auto stage_store = [&](int b) {
        uint4* lh = (uint4*)sbh[b];
        uint4* ll = (uint4*)sbl[b];
        #pragma unroll
        for (int i = 0; i < 4; ++i) {
            lh[tid + i * NTHREADS] = th[i];
            ll[tid + i * NTHREADS] = tl[i];
        }
    };
    stage_load(0);
    stage_store(0);
    __syncthreads();
#endif

    float p0[8], p1[8];
    #pragma unroll
    for (int r = 0; r < 8; ++r) { p0[r] = 0.f; p1[r] = 0.f; }

    for (int t = 0; t < NTILES; ++t) {
        const int k  = t >> 4;
        const int d0 = (t & 15) << 4;
        const bool more = (t + 1) < NTILES;

#if defined(ASYNC_STAGE)
        if (more) stage(t + 1, (t + 1) & 1);  // async, in flight under WMMAs
#else
        if (more) stage_load(t + 1);          // in flight under the WMMAs
#endif

        // ---- accumulators init with -u_k[d] (folds linear term in fp32)
        float nu = negu[(k << 8) + d0 + lm];
        v8f C0, C1;
        #pragma unroll
        for (int r = 0; r < 8; ++r) { C0[r] = nu; C1[r] = nu; }

        // ---- Y += Xh*Sh + Xh*Sl + Xl*Sh over e=0..255 ; two row tiles share B
        const __bf16* bh = sbh[t & 1] + lm * 256 + hi * 8;
        const __bf16* bl = sbl[t & 1] + lm * 256 + hi * 8;
        #pragma unroll
        for (int es = 0; es < 8; ++es) {
            union U { uint4 u[2]; v16bf v; } Bh, Bl;
            Bh.u[0] = *(const uint4*)(bh + es * 32);
            Bh.u[1] = *(const uint4*)(bh + es * 32 + 16);
            Bl.u[0] = *(const uint4*)(bl + es * 32);
            Bl.u[1] = *(const uint4*)(bl + es * 32 + 16);
            C0 = __builtin_amdgcn_wmma_f32_16x16x32_bf16(
                     false, Ah0[es], false, Bh.v, (short)0, C0, false, false);
            C1 = __builtin_amdgcn_wmma_f32_16x16x32_bf16(
                     false, Ah1[es], false, Bh.v, (short)0, C1, false, false);
            C0 = __builtin_amdgcn_wmma_f32_16x16x32_bf16(
                     false, Ah0[es], false, Bl.v, (short)0, C0, false, false);
            C1 = __builtin_amdgcn_wmma_f32_16x16x32_bf16(
                     false, Ah1[es], false, Bl.v, (short)0, C1, false, false);
            C0 = __builtin_amdgcn_wmma_f32_16x16x32_bf16(
                     false, Al0[es], false, Bh.v, (short)0, C0, false, false);
            C1 = __builtin_amdgcn_wmma_f32_16x16x32_bf16(
                     false, Al1[es], false, Bh.v, (short)0, C1, false, false);
        }

        // ---- fused row-dot: partial += Y[m][d0+lm] * X[m][d0+lm]
        const float* xc0 = X + (size_t)(rb0 + hi * 8) * DDIM + d0 + lm;
        const float* xc1 = xc0 + (size_t)16 * DDIM;
        #pragma unroll
        for (int r = 0; r < 8; ++r) {
            p0[r] += C0[r] * xc0[(size_t)r * DDIM];
            p1[r] += C1[r] * xc1[(size_t)r * DDIM];
        }

#if !defined(ASYNC_STAGE)
        if (more) stage_store((t + 1) & 1);
#endif

        // ---- cluster boundary: reduce across the 16 column lanes, store t
        if ((t & 15) == 15) {
            #pragma unroll
            for (int r = 0; r < 8; ++r) {
                float a = p0[r], b = p1[r];
                a += __shfl_xor(a, 1, 32);  b += __shfl_xor(b, 1, 32);
                a += __shfl_xor(a, 2, 32);  b += __shfl_xor(b, 2, 32);
                a += __shfl_xor(a, 4, 32);  b += __shfl_xor(b, 4, 32);
                a += __shfl_xor(a, 8, 32);  b += __shfl_xor(b, 8, 32);
                p0[r] = a; p1[r] = b;
            }
            if (lm == 0) {  // lane 0 -> rows m, lane 16 -> rows m+8
                #pragma unroll
                for (int r = 0; r < 8; ++r) {
                    ts[wave * ROWS_PER_WAVE + hi * 8 + r][k]      = p0[r];
                    ts[wave * ROWS_PER_WAVE + 16 + hi * 8 + r][k] = p1[r];
                }
            }
            #pragma unroll
            for (int r = 0; r < 8; ++r) { p0[r] = 0.f; p1[r] = 0.f; }
        }

        async_drain();     // (async path) our slab writes for t+1 are in LDS
        __syncthreads();
    }

    // ---- finalize: mahal = t + c_k; q = (1/(1+sqrt(mahal))) normalized
    {
        const int row = tid;                   // 128 rows, 1 thread each
        float num[KC];
        float s = 0.f;
        #pragma unroll
        for (int k = 0; k < KC; ++k) {
            float m  = ts[row][k] + cvec[k];
            float md = sqrtf(fmaxf(m, 0.f));
            num[k] = 1.0f / (1.0f + md);
            s += num[k];
        }
        float inv = 1.0f / s;
        float* out = q + (size_t)(blockIdx.x * ROWS_PER_WG + row) * KC;
        #pragma unroll
        for (int k = 0; k < KC; ++k) out[k] = num[k] * inv;
    }
}

// ---------------------------------------------------------------------------
extern "C" void kernel_launch(void* const* d_in, const int* in_sizes, int n_in,
                              void* d_out, int out_size, void* d_ws, size_t ws_size,
                              hipStream_t stream) {
    (void)in_sizes; (void)n_in; (void)out_size; (void)ws_size;
    const float* X  = (const float*)d_in[0];   // [16384, 256]
    const float* mu = (const float*)d_in[1];   // [32, 256]
    const float* S  = (const float*)d_in[2];   // [32, 256, 256]
    float* q = (float*)d_out;                  // [16384, 32]

    char* ws = (char*)d_ws;
    const size_t nST = (size_t)KC * DDIM * DDIM;           // 2,097,152 elems
    __bf16* STh = (__bf16*)ws;                             // 4 MB
    __bf16* STl = (__bf16*)(ws + nST * 2);                 // 4 MB
    float*  negu = (float*)(ws + nST * 4);                 // 32 KB
    float*  cvec = (float*)(ws + nST * 4 + (size_t)KC * DDIM * 4);

    prep_split_transpose<<<(int)(nST / 256), 256, 0, stream>>>(S, STh, STl);
    prep_linear<<<KC, 256, 0, stream>>>(mu, S, negu, cvec);
    mdec_main<<<B_TOTAL / ROWS_PER_WG, NTHREADS, 0, stream>>>(X, STh, STl, negu, cvec, q);
}
